// GraphSageLSTMConv_80865644249293
// MI455X (gfx1250) — matrix operand order repaired and checked by reading, not compiled, not run on hardware
//
#include <hip/hip_runtime.h>
#include <hip/hip_bf16.h>

typedef __bf16 bf16;
typedef __attribute__((ext_vector_type(16))) __bf16 v16bf;
typedef __attribute__((ext_vector_type(8)))  float  v8f;

#define O_NODES 10000
#define T_EDGES 50000
#define DIN 128
#define HID 512
#define DOUTC 128

// ---------------------------------------------------------------------------
// WMMA helpers (CDNA5 wave32, v_wmma_f32_16x16x32_bf16)
//
// A fragment: 16x32 (MxK) bf16, row-major source with leading dim `ld`.
//   lane 0-15 : row = lane,     K = 0..7  then 16..23
//   lane 16-31: row = lane-16,  K = 8..15 then 24..31
// B fragment: 32x16 (KxN) bf16. B = W^T  =>  B[k][n] = W[n][k], so a lane
// reads two 8-bf16 runs (K as for A) from row-major W at row = N-index.
// ---------------------------------------------------------------------------

union frag_cvt { uint4 u[2]; v16bf v; };

__device__ __forceinline__ v16bf load_frag_a(const bf16* base, int ld, int lane) {
  const int row = lane & 15;
  const int k0  = (lane & 16) ? 8 : 0;
  const bf16* p = base + (size_t)row * ld + k0;
  frag_cvt c;
  c.u[0] = *(const uint4*)(p);
  c.u[1] = *(const uint4*)(p + 16);
  return c.v;
}

__device__ __forceinline__ v16bf load_frag_b(const bf16* base, int ld, int lane) {
  const int col = lane & 15;
  const int k0  = (lane & 16) ? 8 : 0;
  const bf16* p = base + (size_t)col * ld + k0;
  frag_cvt c;
  c.u[0] = *(const uint4*)(p);
  c.u[1] = *(const uint4*)(p + 16);
  return c.v;
}

__device__ __forceinline__ v8f wmma_bf16(v16bf a, v16bf b, v8f c) {
  return __builtin_amdgcn_wmma_f32_16x16x32_bf16(false, a, false, b,
                                                 (short)0, c, false, false);
}

__device__ __forceinline__ v8f vzero8() {
  v8f z = {0.f, 0.f, 0.f, 0.f, 0.f, 0.f, 0.f, 0.f};
  return z;
}

__device__ __forceinline__ float sigmoid_f(float x) {
  return 1.0f / (1.0f + __expf(-x));
}

// --- CDNA5 async global<->LDS copies (ASYNCcnt) ----------------------------
// LDS operand = low 32 bits of the generic (flat) address of a __shared__
// object (ISA 10.2: flat LDS address = addr[31:0]).
__device__ __forceinline__ unsigned lds_addr32(const void* p) {
  return (unsigned)(size_t)p;
}

__device__ __forceinline__ void async_load_b128(unsigned lds, const void* gsrc) {
  asm volatile("global_load_async_to_lds_b128 %0, %1, off"
               :: "v"(lds), "v"((unsigned long long)(size_t)gsrc)
               : "memory");
}

__device__ __forceinline__ void async_store_b128(void* gdst, unsigned lds) {
  asm volatile("global_store_async_from_lds_b128 %0, %1, off"
               :: "v"((unsigned long long)(size_t)gdst), "v"(lds)
               : "memory");
}

__device__ __forceinline__ void wait_async0() {
  asm volatile("s_wait_asynccnt 0x0" ::: "memory");
}

// ---------------------------------------------------------------------------
// Prep: fp32 -> bf16 weight conversion, table row-0 zeroing
// ---------------------------------------------------------------------------
__global__ void cvt_f32_bf16_kernel(const float* __restrict__ src,
                                    bf16* __restrict__ dst, int n) {
  int i = blockIdx.x * blockDim.x + threadIdx.x;
  if (i < n) dst[i] = (bf16)src[i];
}

__global__ void zero_bf16_kernel(bf16* __restrict__ dst, int n) {
  int i = blockIdx.x * blockDim.x + threadIdx.x;
  if (i < n) dst[i] = (bf16)0.0f;
}

// ---------------------------------------------------------------------------
// Kernel 1: edge MLP.  16 edges per block, 256 threads (8 waves).
// ---------------------------------------------------------------------------
__global__ __launch_bounds__(256) void edge_mlp_kernel(
    const float* __restrict__ obj_vecs, const float* __restrict__ pred_vecs,
    const int* __restrict__ edges,
    const bf16* __restrict__ W1bf, const float* __restrict__ b1,
    const bf16* __restrict__ W2bf, const float* __restrict__ b2,
    bf16* __restrict__ tablebf, float* __restrict__ newp_out) {
  __shared__ bf16 Xs[16][392];   // 16 x 384, padded
  __shared__ bf16 Ms[16][520];   // 16 x 512, padded

  const int tid  = threadIdx.x;
  const int lane = tid & 31;
  const int w    = tid >> 5;
  const int e0   = blockIdx.x * 16;

  // ---- gather [obj[s] | pred | obj[o]] -> bf16 LDS (fp32->bf16 convert) ----
  {
    const int slot = tid >> 4;
    const int sub  = tid & 15;
    const int e    = e0 + slot;
    const int s    = edges[2 * e];
    const int o    = edges[2 * e + 1];
#pragma unroll
    for (int j = 0; j < 6; ++j) {
      const int c4  = sub * 6 + j;
      const int col = c4 * 4;
      const float* src;
      int cc;
      if (col < 128)      { src = obj_vecs  + (size_t)s * 128; cc = col;       }
      else if (col < 256) { src = pred_vecs + (size_t)e * 128; cc = col - 128; }
      else                { src = obj_vecs  + (size_t)o * 128; cc = col - 256; }
      const float4 v = *(const float4*)(src + cc);
      bf16* d = &Xs[slot][col];
      d[0] = (bf16)v.x; d[1] = (bf16)v.y; d[2] = (bf16)v.z; d[3] = (bf16)v.w;
    }
  }
  __syncthreads();

  // ---- stage 1: M=16 N=512 K=384 ------------------------------------------
  {
    v8f acc[4];
#pragma unroll
    for (int t = 0; t < 4; ++t) acc[t] = vzero8();
    for (int kc = 0; kc < 12; ++kc) {
      const v16bf a = load_frag_a(&Xs[0][kc * 32], 392, lane);
#pragma unroll
      for (int t = 0; t < 4; ++t) {
        const bf16* bp = W1bf + (size_t)((w * 4 + t) * 16) * 384 + kc * 32;
        acc[t] = wmma_bf16(a, load_frag_b(bp, 384, lane), acc[t]);
      }
    }
#pragma unroll
    for (int t = 0; t < 4; ++t) {
      const int col   = (w * 4 + t) * 16 + (lane & 15);
      const float bia = b1[col];
      const int rbase = (lane & 16) ? 8 : 0;
#pragma unroll
      for (int v = 0; v < 8; ++v) {
        float val = fmaxf(acc[t][v] + bia, 0.0f);
        Ms[rbase + v][col] = (bf16)val;
      }
    }
  }
  __syncthreads();

  // ---- stage 2: M=16 N=1152 K=512 -----------------------------------------
  {
    v8f acc[9];
#pragma unroll
    for (int t = 0; t < 9; ++t) acc[t] = vzero8();
    for (int kc = 0; kc < 16; ++kc) {
      const v16bf a = load_frag_a(&Ms[0][kc * 32], 520, lane);
#pragma unroll
      for (int t = 0; t < 9; ++t) {
        const int ntile = w * 9 + t;
        const bf16* bp  = W2bf + (size_t)(ntile * 16) * 512 + kc * 32;
        acc[t] = wmma_bf16(a, load_frag_b(bp, 512, lane), acc[t]);
      }
    }
#pragma unroll
    for (int t = 0; t < 9; ++t) {
      const int ntile = w * 9 + t;
      const int col   = ntile * 16 + (lane & 15);
      const float bia = b2[col];
      const int rbase = (lane & 16) ? 8 : 0;
#pragma unroll
      for (int v = 0; v < 8; ++v) {
        const int e = e0 + rbase + v;
        float val = fmaxf(acc[t][v] + bia, 0.0f);
        if (col < 512) {
          tablebf[(size_t)(1 + e) * 512 + col] = (bf16)val;           // new_s
        } else if (col < 640) {
          newp_out[(size_t)e * 128 + (col - 512)] = val;              // new_p
        } else {
          tablebf[(size_t)(1 + T_EDGES + e) * 512 + (col - 640)] = (bf16)val; // new_o
        }
      }
    }
  }
}

// ---------------------------------------------------------------------------
// Kernel 2: masked LSTM over neighbor sequences. 16 nodes per block.
// x_t staged via GLOBAL_LOAD_ASYNC_TO_LDS_B128; h_n written out via
// GLOBAL_STORE_ASYNC_FROM_LDS_B128 (bf16 raw copies, ASYNCcnt-tracked).
// ---------------------------------------------------------------------------
__global__ __launch_bounds__(256) void lstm_kernel(
    const bf16* __restrict__ tablebf, const int* __restrict__ nbr,
    const int* __restrict__ lengths,
    const bf16* __restrict__ Wihbf, const bf16* __restrict__ Whhbf,
    const float* __restrict__ bih, const float* __restrict__ bhh,
    bf16* __restrict__ hn, int Lnbr) {
  __shared__ bf16  Xs[16][520];      // x_t, bf16
  __shared__ bf16  Hs[2][16][520];   // h, double-buffered bf16
  __shared__ float Cs[16][512];      // c, fp32
  __shared__ int   len_s[16];
  __shared__ int   lmax_s;

  const int tid  = threadIdx.x;
  const int lane = tid & 31;
  const int w    = tid >> 5;
  const int n0   = blockIdx.x * 16;

  if (tid < 16) len_s[tid] = lengths[n0 + tid];
  for (int idx = tid; idx < 16 * 512; idx += 256) {
    const int m = idx >> 9, j = idx & 511;
    Hs[0][m][j] = (bf16)0.0f;
    Cs[m][j]    = 0.0f;
  }
  __syncthreads();
  if (tid == 0) {
    int lm = 0;
    for (int i = 0; i < 16; ++i) lm = (len_s[i] > lm) ? len_s[i] : lm;
    lmax_s = lm;
  }
  __syncthreads();
  const int Lmax = lmax_s;
  const int slot = tid >> 4;
  const int sub  = tid & 15;

  for (int t = 0; t < Lmax; ++t) {
    // ---- async gather of x_t rows from the table (row 0 is zeros) ---------
    {
      const int idx = nbr[(size_t)(n0 + slot) * Lnbr + t];
      const bf16* src = tablebf + (size_t)idx * 512;
      const unsigned ldsrow = lds_addr32(&Xs[slot][0]);
#pragma unroll
      for (int q = 0; q < 4; ++q) {
        const int off = sub * 32 + q * 8;            // bf16 elements
        async_load_b128(ldsrow + off * 2, src + off);
      }
      // prefetch next step's source row while this step computes
      if (t + 1 < Lmax) {
        const int idx2 = nbr[(size_t)(n0 + slot) * Lnbr + t + 1];
        __builtin_prefetch(tablebf + (size_t)idx2 * 512 + sub * 32, 0, 1);
      }
      wait_async0();
    }
    __syncthreads();
    const int cur = t & 1, nxt = cur ^ 1;

#pragma unroll
    for (int js = 0; js < 2; ++js) {
      const int j0 = w * 64 + js * 32;
      v8f acc[4][2];
#pragma unroll
      for (int g = 0; g < 4; ++g)
#pragma unroll
        for (int nt = 0; nt < 2; ++nt) acc[g][nt] = vzero8();

      // x @ Wih^T
      for (int kc = 0; kc < 16; ++kc) {
        const v16bf a = load_frag_a(&Xs[0][kc * 32], 520, lane);
#pragma unroll
        for (int g = 0; g < 4; ++g)
#pragma unroll
          for (int nt = 0; nt < 2; ++nt) {
            const bf16* bp =
                Wihbf + (size_t)(g * 512 + j0 + nt * 16) * 512 + kc * 32;
            acc[g][nt] = wmma_bf16(a, load_frag_b(bp, 512, lane), acc[g][nt]);
          }
      }
      // h @ Whh^T
      for (int kc = 0; kc < 16; ++kc) {
        const v16bf a = load_frag_a(&Hs[cur][0][kc * 32], 520, lane);
#pragma unroll
        for (int g = 0; g < 4; ++g)
#pragma unroll
          for (int nt = 0; nt < 2; ++nt) {
            const bf16* bp =
                Whhbf + (size_t)(g * 512 + j0 + nt * 16) * 512 + kc * 32;
            acc[g][nt] = wmma_bf16(a, load_frag_b(bp, 512, lane), acc[g][nt]);
          }
      }

      // elementwise LSTM cell with mask
#pragma unroll
      for (int nt = 0; nt < 2; ++nt) {
        const int j  = j0 + nt * 16 + (lane & 15);
        const float bI = bih[j]        + bhh[j];
        const float bF = bih[512 + j]  + bhh[512 + j];
        const float bG = bih[1024 + j] + bhh[1024 + j];
        const float bO = bih[1536 + j] + bhh[1536 + j];
        const int rbase = (lane & 16) ? 8 : 0;
#pragma unroll
        for (int v = 0; v < 8; ++v) {
          const int m  = rbase + v;
          const float gi = sigmoid_f(acc[0][nt][v] + bI);
          const float gf = sigmoid_f(acc[1][nt][v] + bF);
          const float gg = tanhf(acc[2][nt][v] + bG);
          const float go = sigmoid_f(acc[3][nt][v] + bO);
          const float c  = Cs[m][j];
          const float c2 = gf * c + gi * gg;
          const float h2 = go * tanhf(c2);
          const bool  mk = t < len_s[m];
          Cs[m][j] = mk ? c2 : c;
          const float hold = (float)Hs[cur][m][j];
          Hs[nxt][m][j] = (bf16)(mk ? h2 : hold);
        }
      }
    }
    __syncthreads();
  }

  // ---- async write-out of final h (bf16, raw LDS->global) ------------------
  {
    const int fin = Lmax & 1;
    const unsigned ldsrow = lds_addr32(&Hs[fin][slot][0]);
    bf16* dst = hn + (size_t)(n0 + slot) * 512;
#pragma unroll
    for (int q = 0; q < 4; ++q) {
      const int off = sub * 32 + q * 8;
      async_store_b128(dst + off, ldsrow + off * 2);
    }
    wait_async0();
  }
}

// ---------------------------------------------------------------------------
// Kernel 3: fused output projection. 16 nodes per block.
//   prev = obj @ Wproj^T + bproj
//   out  = h_n @ Wout[:, :512]^T + prev @ Wout[:, 512:]^T + bout
// h_n (bf16) staged via async global->LDS copies.
// ---------------------------------------------------------------------------
__global__ __launch_bounds__(256) void out_proj_kernel(
    const float* __restrict__ obj_vecs, const bf16* __restrict__ hn,
    const bf16* __restrict__ Wprojbf, const float* __restrict__ bproj,
    const bf16* __restrict__ Woutbf, const float* __restrict__ bout,
    float* __restrict__ out) {
  __shared__ bf16 Os[16][136];   // obj tile  (16x128, padded)
  __shared__ bf16 Hs[16][520];   // h_n tile  (16x512, padded)
  __shared__ bf16 Ps[16][520];   // prev tile (16x512, padded)

  const int tid  = threadIdx.x;
  const int lane = tid & 31;
  const int w    = tid >> 5;
  const int n0   = blockIdx.x * 16;
  const int slot = tid >> 4;
  const int sub  = tid & 15;

  // async raw copy of h_n rows into LDS
  {
    const bf16* src = hn + (size_t)(n0 + slot) * 512;
    const unsigned ldsrow = lds_addr32(&Hs[slot][0]);
#pragma unroll
    for (int q = 0; q < 4; ++q) {
      const int off = sub * 32 + q * 8;
      async_load_b128(ldsrow + off * 2, src + off);
    }
  }
  // obj tile needs fp32->bf16 conversion: regular path
  for (int idx = tid; idx < 16 * 128; idx += 256) {
    const int m = idx >> 7, j = idx & 127;
    Os[m][j] = (bf16)obj_vecs[(size_t)(n0 + m) * 128 + j];
  }
  wait_async0();
  __syncthreads();

  // stage 1: prev(16x512) = obj(16x128) @ Wproj^T
  {
    v8f acc[4];
#pragma unroll
    for (int t = 0; t < 4; ++t) acc[t] = vzero8();
    for (int kc = 0; kc < 4; ++kc) {
      const v16bf a = load_frag_a(&Os[0][kc * 32], 136, lane);
#pragma unroll
      for (int t = 0; t < 4; ++t) {
        const bf16* bp = Wprojbf + (size_t)((w * 4 + t) * 16) * 128 + kc * 32;
        acc[t] = wmma_bf16(a, load_frag_b(bp, 128, lane), acc[t]);
      }
    }
#pragma unroll
    for (int t = 0; t < 4; ++t) {
      const int col   = (w * 4 + t) * 16 + (lane & 15);
      const float bia = bproj[col];
      const int rbase = (lane & 16) ? 8 : 0;
#pragma unroll
      for (int v = 0; v < 8; ++v) Ps[rbase + v][col] = (bf16)(acc[t][v] + bia);
    }
  }
  __syncthreads();

  // stage 2: out(16x128) = Hs @ WoutA^T + Ps @ WoutB^T + bout
  {
    v8f acc = vzero8();
    for (int kc = 0; kc < 16; ++kc) {
      const v16bf a  = load_frag_a(&Hs[0][kc * 32], 520, lane);
      const bf16* bp = Woutbf + (size_t)(w * 16) * 1024 + kc * 32;
      acc = wmma_bf16(a, load_frag_b(bp, 1024, lane), acc);
    }
    for (int kc = 0; kc < 16; ++kc) {
      const v16bf a  = load_frag_a(&Ps[0][kc * 32], 520, lane);
      const bf16* bp = Woutbf + (size_t)(w * 16) * 1024 + 512 + kc * 32;
      acc = wmma_bf16(a, load_frag_b(bp, 1024, lane), acc);
    }
    const int col   = w * 16 + (lane & 15);
    const float bia = bout[col];
    const int rbase = (lane & 16) ? 8 : 0;
#pragma unroll
    for (int v = 0; v < 8; ++v)
      out[(size_t)(n0 + rbase + v) * 128 + col] = acc[v] + bia;
  }
}

// ---------------------------------------------------------------------------
// Host launcher
// ---------------------------------------------------------------------------
extern "C" void kernel_launch(void* const* d_in, const int* in_sizes, int n_in,
                              void* d_out, int out_size, void* d_ws,
                              size_t ws_size, hipStream_t stream) {
  const float* obj    = (const float*)d_in[0];
  const float* pred   = (const float*)d_in[1];
  const int*   edges  = (const int*)d_in[2];
  const int*   nbr    = (const int*)d_in[3];
  const int*   length = (const int*)d_in[4];
  const float* W1     = (const float*)d_in[5];
  const float* b1     = (const float*)d_in[6];
  const float* W2     = (const float*)d_in[7];
  const float* b2     = (const float*)d_in[8];
  const float* Wih    = (const float*)d_in[9];
  const float* Whh    = (const float*)d_in[10];
  const float* bih    = (const float*)d_in[11];
  const float* bhh    = (const float*)d_in[12];
  const float* Wproj  = (const float*)d_in[13];
  const float* bproj  = (const float*)d_in[14];
  const float* Wout   = (const float*)d_in[15];
  const float* bout   = (const float*)d_in[16];
  const int Lnbr = in_sizes[3] / O_NODES;

  char* ws = (char*)d_ws;
  size_t off = 0;
  bf16* W1bf    = (bf16*)(ws + off); off += (size_t)512 * 384 * 2;
  bf16* W2bf    = (bf16*)(ws + off); off += (size_t)1152 * 512 * 2;
  bf16* Wihbf   = (bf16*)(ws + off); off += (size_t)2048 * 512 * 2;
  bf16* Whhbf   = (bf16*)(ws + off); off += (size_t)2048 * 512 * 2;
  bf16* Wprojbf = (bf16*)(ws + off); off += (size_t)512 * 128 * 2;
  bf16* Woutbf  = (bf16*)(ws + off); off += (size_t)128 * 1024 * 2;
  bf16* tablebf = (bf16*)(ws + off); off += (size_t)(1 + 2 * T_EDGES) * 512 * 2;
  bf16* hn      = (bf16*)(ws + off); off += (size_t)O_NODES * 512 * 2;
  (void)ws_size; (void)n_in; (void)out_size;

  auto cvt = [&](const float* s, bf16* d, int n) {
    cvt_f32_bf16_kernel<<<(n + 255) / 256, 256, 0, stream>>>(s, d, n);
  };
  cvt(W1, W1bf, 512 * 384);
  cvt(W2, W2bf, 1152 * 512);
  cvt(Wih, Wihbf, 2048 * 512);
  cvt(Whh, Whhbf, 2048 * 512);
  cvt(Wproj, Wprojbf, 512 * 128);
  cvt(Wout, Woutbf, 128 * 1024);
  zero_bf16_kernel<<<2, 256, 0, stream>>>(tablebf, 512);  // table row 0 = 0

  float* new_obj = (float*)d_out;                            // 10000 x 128
  float* new_p   = (float*)d_out + (size_t)O_NODES * DOUTC;  // 50000 x 128

  edge_mlp_kernel<<<T_EDGES / 16, 256, 0, stream>>>(
      obj, pred, edges, W1bf, b1, W2bf, b2, tablebf, new_p);

  lstm_kernel<<<O_NODES / 16, 256, 0, stream>>>(
      tablebf, nbr, length, Wihbf, Whhbf, bih, bhh, hn, Lnbr);

  out_proj_kernel<<<O_NODES / 16, 256, 0, stream>>>(
      obj, hn, Wprojbf, bproj, Woutbf, bout, new_obj);
}
